// SelfAttentionV2_798863917454
// MI455X (gfx1250) — matrix operand order repaired and checked by reading, not compiled, not run on hardware
//
#include <hip/hip_runtime.h>

// Self-attention (B=4, S=2048, H=1024, single head) for gfx1250 / MI455X.
// ~120 GFLOP total; bf16 WMMA (f32 accum) keeps matrix time (~24us) balanced
// against 23.3 TB/s HBM (~12us). All GEMM inner loops are pure bf16:
//   kernel0: one-pass f32->bf16 convert of X and W (removes cvt from hot loops)
//   kernels 1/2: B tiles double-buffered via GLOBAL_LOAD_ASYNC_TO_LDS_B128
//   kernel 4:    B tiles double-buffered via TDM tensor_load_to_lds (TENSORcnt)
// Workspace (102 MB): Qb(16M) Kb(16M) Vt(16M) Sc(32M) Xb(16M) Wb(6M).

#define HID 1024
#define O3H 3072
#define SEQ 2048
#define NB  4
#define ROWS (NB*SEQ)          // 8192 flattened (b,s) rows
#define LDS_STRIDE 72          // C-tile staging: 64 cols + 8 pad floats
#define BSTRIDE_BF 80          // bf16 B-tile row stride bytes (64 data + 16 pad)

typedef __attribute__((ext_vector_type(16))) __bf16 v16bf;
typedef __attribute__((ext_vector_type(8)))  __bf16 v8bf;
typedef __attribute__((ext_vector_type(8)))  float  v8f;
typedef __attribute__((ext_vector_type(4)))  float  v4f;
typedef unsigned int u32x4 __attribute__((ext_vector_type(4)));
typedef int          i32x4 __attribute__((ext_vector_type(4)));
typedef int          i32x8 __attribute__((ext_vector_type(8)));

#if __has_builtin(__builtin_amdgcn_tensor_load_to_lds)
#define HAVE_TDM 1
#else
#define HAVE_TDM 0
#endif
static constexpr bool kHaveTdm = (HAVE_TDM != 0);

// ---- CDNA5 async copy: global -> LDS, 16B per lane, tracked by ASYNCcnt ---
static __device__ __forceinline__ void async_b128(const void* g, void* l) {
  unsigned ldsoff = (unsigned)(unsigned long long)l;  // flat[31:0] == LDS byte addr
  asm volatile("global_load_async_to_lds_b128 %0, %1, off"
               :: "v"(ldsoff), "v"(g) : "memory");
}
static __device__ __forceinline__ void wait_async0() {
  asm volatile("s_wait_asynccnt 0x0" ::: "memory");
}
static __device__ __forceinline__ void wait_tensor0() {
#if __has_builtin(__builtin_amdgcn_s_wait_tensorcnt)
  __builtin_amdgcn_s_wait_tensorcnt(0);
#else
  asm volatile("s_wait_tensorcnt 0x0" ::: "memory");
#endif
}

// ---- TDM: move one 64-row x 64-byte bf16 tile (row stride ld_elems) into
// LDS at 80B/row (D# pad: 16 DWORDs data + 4 DWORDs pad). Issued per-wave.
static __device__ __forceinline__ void tdm_load_tile(const void* gsrc, void* ldst,
                                                     int ld_elems) {
#if HAVE_TDM
  unsigned long long ga = (unsigned long long)gsrc;
  unsigned lds = (unsigned)(unsigned long long)ldst;
  u32x4 g0;
  g0[0] = 1u;                                          // count=1 (valid user D#)
  g0[1] = lds;                                         // lds_addr
  g0[2] = (unsigned)(ga & 0xffffffffu);                // global_addr[31:0]
  g0[3] = (unsigned)((ga >> 32) & 0x01ffffffu) | (2u << 30);  // [56:32] | type=2
  i32x8 g1;
  g1[0] = (1 << 16) | (1 << 20) | (3 << 22) | (3 << 25);  // 2B elems, pad 16+4 DW
  g1[1] = (ld_elems & 0xffff) << 16;                   // tensor_dim0[15:0]
  g1[2] = ((ld_elems >> 16) & 0xffff) | (0xffff << 16);// dim0[31:16] | dim1 lo
  g1[3] = 32 << 16;                                    // dim1 hi=0 | tile_dim0=32
  g1[4] = 64;                                          // tile_dim1=64, tile_dim2=0
  g1[5] = ld_elems;                                    // tensor_dim0_stride[31:0]
  g1[6] = 0;                                           // stride hi | dim1_stride lo
  g1[7] = 0;
  i32x4 z4 = {0, 0, 0, 0};
#if __clang_major__ >= 23
  i32x8 z8 = {0, 0, 0, 0, 0, 0, 0, 0};
  __builtin_amdgcn_tensor_load_to_lds(g0, g1, z4, z4, z8, 0);
#else
  __builtin_amdgcn_tensor_load_to_lds(g0, g1, z4, z4, 0);
#endif
#else
  (void)gsrc; (void)ldst; (void)ld_elems;
#endif
}

static __device__ __forceinline__ v8f wmma_bf16(v16bf a, v16bf b, v8f c) {
  return __builtin_amdgcn_wmma_f32_16x16x32_bf16(false, a, false, b, (short)0, c,
                                                 false, false);
}

// A (16x32 bf16): lane: m = L&15, hi = L>>4; element e holds
// k = (e<8 ? e : e+8) + 8*hi -> two 8-elem runs at kb+8*hi and kb+16+8*hi.
static __device__ __forceinline__ v16bf combine_a(v8bf x0, v8bf x1) {
  v16bf a;
#pragma unroll
  for (int i = 0; i < 8; ++i) { a[i] = x0[i]; a[8 + i] = x1[i]; }
  return a;
}
// B (32x16): lane: n = L&15, hi = L>>4; element e -> k = (e<8 ? e : e+8) + 8*hi.
static __device__ __forceinline__ v16bf frag_b_lds_bf16(const char* p) {
  const v8bf* q = (const v8bf*)p;
  v8bf x0 = q[0], x1 = q[2];
  v16bf b;
#pragma unroll
  for (int i = 0; i < 8; ++i) { b[i] = x0[i]; b[8 + i] = x1[i]; }
  return b;
}

// ---- kernel 0: elementwise f32 -> bf16 (8 elems/thread) -------------------
__global__ __launch_bounds__(256) void cvt_kernel(const float* __restrict__ src,
                                                  __bf16* __restrict__ dst) {
  const size_t i8 = ((size_t)blockIdx.x * 256 + threadIdx.x) * 8;
  v4f x0 = ((const v4f*)(src + i8))[0];
  v4f x1 = ((const v4f*)(src + i8))[1];
  v8bf o;
#pragma unroll
  for (int i = 0; i < 4; ++i) { o[i] = (__bf16)x0[i]; o[4 + i] = (__bf16)x1[i]; }
  *(v8bf*)(dst + i8) = o;
}

// ---- generic bf16 GEMM body ----------------------------------------------
// 128x64 block tile (8 waves x 16 rows), K step 32. A row-major (lda);
// B rows are output columns (ldb). Result (scaled) left staged in cstage.
template <bool USE_TDM>
static __device__ __forceinline__ void gemm_bf16_tile(
    const __bf16* __restrict__ Abase, int lda, const __bf16* __restrict__ Bbase,
    int ldb, int Kdim, float scale, float* cstage, char (*bstage)[64 * BSTRIDE_BF],
    int tid, int r0, int n0) {
  const int lane = tid & 31, w = tid >> 5;
  const int m16 = lane & 15, hi = lane >> 4;

  v8f acc[4];
#pragma unroll
  for (int s = 0; s < 4; ++s)
#pragma unroll
    for (int i = 0; i < 8; ++i) acc[s][i] = 0.0f;

  const int bn = tid >> 2, bseg = tid & 3;  // 64 rows x 4 x 16B = full tile
  const __bf16* bsrc = Bbase + (size_t)(n0 + bn) * ldb + bseg * 8;
  char* bdst0 = &bstage[0][bn * BSTRIDE_BF + bseg * 16];
  char* bdst1 = &bstage[1][bn * BSTRIDE_BF + bseg * 16];
  const __bf16* btile = Bbase + (size_t)n0 * ldb;     // TDM tile origin
  const __bf16* Arow = Abase + (size_t)(r0 + m16) * lda + hi * 8;

  if constexpr (USE_TDM && kHaveTdm) {
    if (w == 0) tdm_load_tile(btile, &bstage[0][0], ldb);
  } else {
    async_b128(bsrc, bdst0);
  }
  v8bf a0 = ((const v8bf*)Arow)[0], a1 = ((const v8bf*)Arow)[2];
  if constexpr (USE_TDM && kHaveTdm) {
    if (w == 0) wait_tensor0();
  } else {
    wait_async0();
  }
  __syncthreads();

  for (int kb = 0; kb < Kdim; kb += 32) {
    const int cur = (kb >> 5) & 1;
    v8bf na0 = a0, na1 = a1;
    if (kb + 32 < Kdim) {                       // issue next B tile early
      if constexpr (USE_TDM && kHaveTdm) {
        if (w == 0) tdm_load_tile(btile + kb + 32, &bstage[cur ^ 1][0], ldb);
      } else {
        async_b128(bsrc + kb + 32, cur ? bdst0 : bdst1);
      }
      const __bf16* an = Arow + kb + 32;        // prefetch next A fragment
      na0 = ((const v8bf*)an)[0];
      na1 = ((const v8bf*)an)[2];
    }
    v16bf a = combine_a(a0, a1);
    // pull all four B fragments first so the 8 ds_load_b128 can clause and
    // the scheduler can use partial s_wait_dscnt instead of a drain per WMMA
    v16bf bfr[4];
#pragma unroll
    for (int sub = 0; sub < 4; ++sub)
      bfr[sub] = frag_b_lds_bf16(&bstage[cur][(sub * 16 + m16) * BSTRIDE_BF + hi * 16]);
#pragma unroll
    for (int sub = 0; sub < 4; ++sub)
      acc[sub] = wmma_bf16(a, bfr[sub], acc[sub]);
    a0 = na0; a1 = na1;
    if constexpr (USE_TDM && kHaveTdm) {
      if (w == 0) wait_tensor0();               // next tile resident
    } else {
      wait_async0();
    }
    __syncthreads();                            // all waves done with bstage[cur]
  }
#pragma unroll
  for (int sub = 0; sub < 4; ++sub)
#pragma unroll
    for (int v = 0; v < 8; ++v)
      cstage[(w * 16 + v + 8 * hi) * LDS_STRIDE + sub * 16 + m16] =
          acc[sub][v] * scale;
  __syncthreads();
}

// ---- kernel 1: QKV = Xb*Wb^T + b; emit Q,K bf16 row-major, V transposed --
__global__ __launch_bounds__(256) void qkv_kernel(
    const __bf16* __restrict__ Xb, const __bf16* __restrict__ Wb,
    const float* __restrict__ bias, __bf16* __restrict__ Qb,
    __bf16* __restrict__ Kb, __bf16* __restrict__ Vt) {
  __shared__ __align__(16) char bstage[2][64 * BSTRIDE_BF];  // 10 KB
  __shared__ float cstage[128 * LDS_STRIDE];                 // 36 KB
  const int tid = threadIdx.x;
  const int blockRow = blockIdx.x;   // 64 tiles of 128 rows
  const int n0 = blockIdx.y * 64;    // 48 chunks of 64 output cols
  const int r0 = blockRow * 128 + (tid >> 5) * 16;

  gemm_bf16_tile<false>(Xb, HID, Wb, HID, HID, 1.0f, cstage, bstage, tid, r0, n0);

  if (n0 < 2 * HID) {  // Q or K: row-major bf16, coalesced 16B stores
    __bf16* dst = (n0 < HID) ? Qb : Kb;
    const int nbase = (n0 < HID) ? n0 : n0 - HID;
    const int row = tid >> 1, halfc = (tid & 1) * 32;
    const float* srcp = &cstage[row * LDS_STRIDE + halfc];
    const float* bp = bias + n0 + halfc;
    __bf16* outp = dst + (size_t)(blockRow * 128 + row) * HID + nbase + halfc;
#pragma unroll
    for (int g = 0; g < 4; ++g) {
      v8bf pk;
#pragma unroll
      for (int i = 0; i < 8; ++i)
        pk[i] = (__bf16)(srcp[g * 8 + i] + bp[g * 8 + i]);
      ((v8bf*)outp)[g] = pk;
    }
  } else {  // V: transpose in LDS, store Vt[b][h][s] contiguous along s
    const int c = tid & 63, rseg = tid >> 6;
    const int batch = (blockRow * 128) / SEQ;
    const int s0 = (blockRow * 128) % SEQ + rseg * 32;
    const int h = n0 - 2 * HID + c;
    const float bv = bias[n0 + c];
    __bf16* outp = Vt + (size_t)batch * HID * SEQ + (size_t)h * SEQ + s0;
#pragma unroll
    for (int g = 0; g < 4; ++g) {
      v8bf pk;
#pragma unroll
      for (int i = 0; i < 8; ++i)
        pk[i] = (__bf16)(cstage[(rseg * 32 + g * 8 + i) * LDS_STRIDE + c] + bv);
      ((v8bf*)outp)[g] = pk;
    }
  }
}

// ---- kernel 2: Sc = (Q K^T)/sqrt(H), bf16 (async-LDS path) ---------------
__global__ __launch_bounds__(256) void scores_kernel(
    const __bf16* __restrict__ Qb, const __bf16* __restrict__ Kb,
    __bf16* __restrict__ Sc) {
  __shared__ __align__(16) char bstage[2][64 * BSTRIDE_BF];
  __shared__ float cstage[128 * LDS_STRIDE];
  const int tid = threadIdx.x;
  const int blockRow = blockIdx.x;   // 128 | 2048 -> single batch per block
  const int n0 = blockIdx.y * 64;
  const int r0 = blockRow * 128 + (tid >> 5) * 16;
  const int batch = (blockRow * 128) / SEQ;

  gemm_bf16_tile<false>(Qb, HID, Kb + (size_t)batch * SEQ * HID, HID, HID,
                        0.03125f /* 1/sqrt(1024) */, cstage, bstage, tid, r0, n0);

  const int row = tid >> 1, halfc = (tid & 1) * 32;
  const float* srcp = &cstage[row * LDS_STRIDE + halfc];
  __bf16* outp = Sc + (size_t)(blockRow * 128 + row) * SEQ + n0 + halfc;
#pragma unroll
  for (int g = 0; g < 4; ++g) {
    v8bf pk;
#pragma unroll
    for (int i = 0; i < 8; ++i) pk[i] = (__bf16)srcp[g * 8 + i];
    ((v8bf*)outp)[g] = pk;
  }
}

// ---- kernel 3: in-place row softmax (one wave per 2048-wide row) ---------
__global__ __launch_bounds__(256) void softmax_kernel(__bf16* __restrict__ Sc) {
  const int tid = threadIdx.x, lane = tid & 31, w = tid >> 5;
  const size_t row = (size_t)blockIdx.x * 8 + w;
  __bf16* p = Sc + row * SEQ;

  float vals[64];
#pragma unroll
  for (int i = 0; i < 8; ++i) {
    v8bf c = ((const v8bf*)p)[i * 32 + lane];
#pragma unroll
    for (int j = 0; j < 8; ++j) vals[i * 8 + j] = (float)c[j];
  }
  float mx = -1e30f;
#pragma unroll
  for (int i = 0; i < 64; ++i) mx = fmaxf(mx, vals[i]);
#pragma unroll
  for (int off = 16; off > 0; off >>= 1) mx = fmaxf(mx, __shfl_xor(mx, off, 32));
  float sum = 0.0f;
#pragma unroll
  for (int i = 0; i < 64; ++i) { vals[i] = __expf(vals[i] - mx); sum += vals[i]; }
#pragma unroll
  for (int off = 16; off > 0; off >>= 1) sum += __shfl_xor(sum, off, 32);
  const float inv = 1.0f / sum;
#pragma unroll
  for (int i = 0; i < 8; ++i) {
    v8bf c;
#pragma unroll
    for (int j = 0; j < 8; ++j) c[j] = (__bf16)(vals[i * 8 + j] * inv);
    ((v8bf*)p)[i * 32 + lane] = c;
  }
}

// ---- kernel 4: out = P * V (TDM-staged B tiles) --------------------------
__global__ __launch_bounds__(256) void av_kernel(
    const __bf16* __restrict__ Sc, const __bf16* __restrict__ Vt,
    float* __restrict__ Out) {
  __shared__ __align__(16) char bstage[2][64 * BSTRIDE_BF];
  __shared__ float cstage[128 * LDS_STRIDE];
  const int tid = threadIdx.x;
  const int blockRow = blockIdx.x;
  const int n0 = blockIdx.y * 64;    // h chunk
  const int r0 = blockRow * 128 + (tid >> 5) * 16;
  const int batch = (blockRow * 128) / SEQ;

  gemm_bf16_tile<true>(Sc, SEQ, Vt + (size_t)batch * HID * SEQ, SEQ, SEQ, 1.0f,
                       cstage, bstage, tid, r0, n0);

  const int row = tid >> 1, halfc = (tid & 1) * 32;
  const float* srcp = &cstage[row * LDS_STRIDE + halfc];
  float* outp = Out + (size_t)(blockRow * 128 + row) * HID + n0 + halfc;
#pragma unroll
  for (int g = 0; g < 8; ++g) ((v4f*)outp)[g] = ((const v4f*)srcp)[g];
}

// ---- launch ---------------------------------------------------------------
extern "C" void kernel_launch(void* const* d_in, const int* in_sizes, int n_in,
                              void* d_out, int out_size, void* d_ws, size_t ws_size,
                              hipStream_t stream) {
  const float* X    = (const float*)d_in[0];  // [4,2048,1024]
  const float* W    = (const float*)d_in[1];  // [3072,1024]
  const float* bias = (const float*)d_in[2];  // [3072]
  float* Out = (float*)d_out;                 // [4,2048,1024]

  char* ws = (char*)d_ws;
  __bf16* Qb = (__bf16*)(ws);                          // 16 MB [b][s][h]
  __bf16* Kb = (__bf16*)(ws + ((size_t)16 << 20));     // 16 MB [b][s][h]
  __bf16* Vt = (__bf16*)(ws + ((size_t)32 << 20));     // 16 MB [b][h][s]
  __bf16* Sc = (__bf16*)(ws + ((size_t)48 << 20));     // 32 MB [b][q][k]
  __bf16* Xb = (__bf16*)(ws + ((size_t)80 << 20));     // 16 MB [b][s][h]
  __bf16* Wb = (__bf16*)(ws + ((size_t)96 << 20));     //  6 MB [o][h]

  dim3 blk(256);
  cvt_kernel<<<dim3((ROWS * HID) / (256 * 8)), blk, 0, stream>>>(X, Xb);
  cvt_kernel<<<dim3((O3H * HID) / (256 * 8)), blk, 0, stream>>>(W, Wb);
  qkv_kernel<<<dim3(ROWS / 128, O3H / 64), blk, 0, stream>>>(Xb, Wb, bias, Qb, Kb, Vt);
  scores_kernel<<<dim3(ROWS / 128, SEQ / 64), blk, 0, stream>>>(Qb, Kb, Sc);
  softmax_kernel<<<dim3(ROWS / 8), blk, 0, stream>>>(Sc);
  av_kernel<<<dim3(ROWS / 128, HID / 64), blk, 0, stream>>>(Sc, Vt, Out);
}
